// VectorQuantizer_9912784519235
// MI455X (gfx1250) — matrix operand order repaired and checked by reading, not compiled, not run on hardware
//
#include <hip/hip_runtime.h>

#define NUM_EMB 8192
#define LATENT  128
#define BATCH   32768

typedef __attribute__((ext_vector_type(16))) __bf16 v16bf;
typedef __attribute__((ext_vector_type(8)))  float  v8f;
typedef __attribute__((ext_vector_type(4)))  unsigned int v4u;
typedef __attribute__((ext_vector_type(8)))  int v8i;
typedef __attribute__((ext_vector_type(4)))  int v4i;

struct alignas(16) U128 { unsigned int w[4]; };
struct U256 { U128 a, b; };

// Assemble a 16-element bf16 fragment from two 8-element (16B) groups.
static __device__ inline v16bf load2x8bf(const unsigned short* p0, const unsigned short* p1) {
    U256 u;
    u.a = *reinterpret_cast<const U128*>(p0);
    u.b = *reinterpret_cast<const U128*>(p1);
    return __builtin_bit_cast(v16bf, u);
}

// float -> bf16 with round-to-nearest-even
static __device__ inline unsigned short f2bf(float f) {
    unsigned int x = __float_as_uint(f);
    x += 0x7fffu + ((x >> 16) & 1u);
    return (unsigned short)(x >> 16);
}

// TDM: DMA one 64-row x 128-col bf16 tile (row stride 128) into LDS with
// 16B padding appended per 256B row => 272B LDS row stride.
// D# per CDNA5 ISA 8.3/8.4; 6-arg builtin (clang-23 / therock headers).
static __device__ inline void tdm_load_tile(const unsigned short* gsrc, unsigned ldsOff) {
    unsigned long long ga = (unsigned long long)(uintptr_t)gsrc;
    v4u g0 = {
        1u,                                    // count=1 (valid user descriptor)
        ldsOff,                                // lds_addr (bytes)
        (unsigned)(ga & 0xffffffffu),          // global_addr[31:0]
        (unsigned)((ga >> 32) & 0x01ffffffu)   // global_addr[56:32]
            | 0x80000000u                      // type=2 ("image")
    };
    v8i g1 = {
        (int)0x07510000,      // data_size=2B, pad_enable, pad_interval=64dw, pad_amount=4dw
        (int)(128u << 16),    // tensor_dim0 = 128 (bits 63:48)
        (int)(64u  << 16),    // tensor_dim0 hi=0 | tensor_dim1 = 64 (bits 95:80)
        (int)(128u << 16),    // tensor_dim1 hi=0 | tile_dim0 = 128 (bits 127:112)
        (int)64,              // tile_dim1 = 64, tile_dim2 = 0
        (int)128,             // tensor_dim0_stride = 128 (bits 191:160)
        0, 0                  // stride hi / tensor_dim1_stride = 0
    };
    v4i gz4 = { 0, 0, 0, 0 };           // groups 2/3 unused (2D tile)
    v8i gz8 = { 0, 0, 0, 0, 0, 0, 0, 0 };
    __builtin_amdgcn_tensor_load_to_lds(g0, g1, gz4, gz4, gz8, 0);
}

// ---------------------------------------------------------------------------
// z_e f32 -> bf16 (vectorized, 4 elems/thread)
__global__ __launch_bounds__(256)
void vq_convert_z(const float* __restrict__ z, unsigned short* __restrict__ zbf) {
    int i = blockIdx.x * 256 + threadIdx.x;            // i < BATCH*LATENT/4
    float4 f = reinterpret_cast<const float4*>(z)[i];
    reinterpret_cast<ushort4*>(zbf)[i] =
        make_ushort4(f2bf(f.x), f2bf(f.y), f2bf(f.z), f2bf(f.w));
}

// W f32 -> bf16, plus exact f32 row norms ||w||^2
__global__ __launch_bounds__(128)
void vq_convert_w(const float* __restrict__ W, unsigned short* __restrict__ wbf,
                  float* __restrict__ wnorm) {
    int k = blockIdx.x;
    int j = threadIdx.x;
    float w = W[k * LATENT + j];
    wbf[k * LATENT + j] = f2bf(w);
    float sq = w * w;
    #pragma unroll
    for (int m = 1; m <= 16; m <<= 1) sq += __shfl_xor(sq, m, 32);
    __shared__ float sp[4];
    if ((j & 31) == 0) sp[j >> 5] = sq;
    __syncthreads();
    if (j == 0) wnorm[k] = (sp[0] + sp[1]) + (sp[2] + sp[3]);
}

// ---------------------------------------------------------------------------
// Main kernel. Block = 256 threads (8 waves); each wave owns TWO 16-row A
// tiles (M=32) so every B fragment read from LDS feeds two WMMAs.
// Grid = BATCH/256 blocks. Codebook staged by TDM, double-buffered.
__global__ __launch_bounds__(256)
void vq_main(const unsigned short* __restrict__ zbf,
             const unsigned short* __restrict__ wbf,
             const float* __restrict__ wnorm,
             const float* __restrict__ z_e,
             const float* __restrict__ W,
             float* __restrict__ out_zq,
             float* __restrict__ out_idx,
             float* __restrict__ blockPartial) {
    // two buffers: 64 codewords x 128 dims each, padded stride 136 elems (272B)
    __shared__ unsigned short sW[2][64 * 136];
    __shared__ float sLoss[8];

    const int tid     = threadIdx.x;
    const int wid     = tid >> 5;
    const int lane    = tid & 31;
    const int halfSel = lane >> 4;    // 0: lanes 0-15, 1: lanes 16-31
    const int lane16  = lane & 15;

    const int m0 = (blockIdx.x * 8 + wid) * 32;   // first z-row of this wave

    // A fragments for two 16-row tiles, per ISA A layout:
    // lane<16 holds K {c*32+0..7, c*32+16..23}; lane>=16 holds {+8..15, +24..31}
    v16bf aF[2][4];
    #pragma unroll
    for (int t = 0; t < 2; ++t) {
        const int rowA = m0 + t * 16 + lane16;
        #pragma unroll
        for (int c = 0; c < 4; ++c) {
            const unsigned short* p = zbf + (size_t)rowA * LATENT + c * 32 + halfSel * 8;
            aF[t][c] = load2x8bf(p, p + 16);
        }
    }

    float bestD[2][8];
    int   bestI[2][8];
    #pragma unroll
    for (int t = 0; t < 2; ++t)
        #pragma unroll
        for (int r = 0; r < 8; ++r) { bestD[t][r] = 3.4e38f; bestI[t][r] = 0; }

    // Prologue: kick TDM for stage 0 into buffer 0
    if (wid == 0)
        tdm_load_tile(wbf, (unsigned)(uintptr_t)(void*)&sW[0][0]);

    for (int stage = 0; stage < NUM_EMB / 64; ++stage) {
        const int nBase = stage * 64;
        const int p     = stage & 1;
        __syncthreads();   // everyone done reading buf[p^1] (stage-1 compute)
        if (wid == 0) {
            __builtin_amdgcn_s_wait_tensorcnt(0);   // buf[p] filled
            if (stage + 1 < NUM_EMB / 64)           // prefetch next stage -> buf[p^1]
                tdm_load_tile(wbf + (size_t)(nBase + 64) * LATENT,
                              (unsigned)(uintptr_t)(void*)&sW[p ^ 1][0]);
        }
        __syncthreads();   // publish buf[p] to all waves

        #pragma unroll
        for (int nt = 0; nt < 4; ++nt) {
            v8f acc0 = {};
            v8f acc1 = {};
            #pragma unroll
            for (int c = 0; c < 4; ++c) {
                // B fragment: lane=col; lanes 0-15 hold K c*32+{0..7,16..23},
                // lanes 16-31 hold K c*32+{8..15,24..31} (matches A)
                const unsigned short* bp =
                    &sW[p][(nt * 16 + lane16) * 136 + c * 32 + halfSel * 8];
                v16bf bF = load2x8bf(bp, bp + 16);
                acc0 = __builtin_amdgcn_wmma_f32_16x16x32_bf16(
                    false, aF[0][c], false, bF, (short)0, acc0, false, false);
                acc1 = __builtin_amdgcn_wmma_f32_16x16x32_bf16(
                    false, aF[1][c], false, bF, (short)0, acc1, false, false);
            }
            const int nGlob = nBase + nt * 16 + lane16;
            const float wn  = wnorm[nGlob];
            // C layout: VGPR r = row base+r (lanes 0-15) / base+8+r (lanes 16-31)
            #pragma unroll
            for (int r = 0; r < 8; ++r) {
                float d0 = fmaf(-2.0f, acc0[r], wn);
                if (d0 < bestD[0][r]) { bestD[0][r] = d0; bestI[0][r] = nGlob; }
                float d1 = fmaf(-2.0f, acc1[r], wn);
                if (d1 < bestD[1][r]) { bestD[1][r] = d1; bestI[1][r] = nGlob; }
            }
        }
    }

    // Argmin across the 16 lanes of each half-wave (butterfly; ties -> lower idx)
    #pragma unroll
    for (int t = 0; t < 2; ++t) {
        #pragma unroll
        for (int r = 0; r < 8; ++r) {
            float d = bestD[t][r];
            int   i = bestI[t][r];
            #pragma unroll
            for (int m = 1; m <= 8; m <<= 1) {
                float od = __shfl_xor(d, m, 32);
                int   oi = __shfl_xor(i, m, 32);
                if (od < d || (od == d && oi < i)) { d = od; i = oi; }
            }
            bestD[t][r] = d; bestI[t][r] = i;
        }
    }

    // Write indices (as float, per output dtype)
    if (lane == 0 || lane == 16) {
        #pragma unroll
        for (int t = 0; t < 2; ++t) {
            int base = m0 + t * 16 + halfSel * 8;
            #pragma unroll
            for (int r = 0; r < 8; ++r)
                out_idx[base + r] = (float)bestI[t][r];
        }
    }

    // Gather z_q rows in f32 and accumulate squared-error loss
    float lossAcc = 0.0f;
    #pragma unroll
    for (int t = 0; t < 2; ++t) {
        #pragma unroll
        for (int rr = 0; rr < 16; ++rr) {
            const int srcLane = (rr < 8) ? 0 : 16;
            const int idx = __shfl(bestI[t][rr & 7], srcLane, 32);
            const int row = m0 + t * 16 + rr;
            #pragma unroll
            for (int j = lane; j < LATENT; j += 32) {
                float w = W[(size_t)idx * LATENT + j];
                out_zq[(size_t)row * LATENT + j] = w;
                float dz = w - z_e[(size_t)row * LATENT + j];
                lossAcc = fmaf(dz, dz, lossAcc);
            }
        }
    }
    #pragma unroll
    for (int m = 1; m <= 16; m <<= 1) lossAcc += __shfl_xor(lossAcc, m, 32);
    if (lane == 0) sLoss[wid] = lossAcc;
    __syncthreads();
    if (tid == 0) {
        float s = 0.0f;
        #pragma unroll
        for (int w = 0; w < 8; ++w) s += sLoss[w];   // fixed order -> deterministic
        blockPartial[blockIdx.x] = s;
    }
}

// ---------------------------------------------------------------------------
__global__ void vq_finalize(const float* __restrict__ partial, float* __restrict__ outLoss) {
    if (threadIdx.x == 0 && blockIdx.x == 0) {
        float s = 0.0f;
        for (int b = 0; b < BATCH / 256; ++b) s += partial[b];  // fixed order
        float mean = s / (float)((size_t)BATCH * LATENT);
        outLoss[0] = mean;   // codebook loss
        outLoss[1] = mean;   // commitment loss (numerically identical)
    }
}

// ---------------------------------------------------------------------------
extern "C" void kernel_launch(void* const* d_in, const int* in_sizes, int n_in,
                              void* d_out, int out_size, void* d_ws, size_t ws_size,
                              hipStream_t stream) {
    (void)in_sizes; (void)n_in; (void)out_size; (void)ws_size;

    const float* z_e = (const float*)d_in[0];   // [BATCH, LATENT] f32
    const float* W   = (const float*)d_in[1];   // [NUM_EMB, LATENT] f32
    float* out = (float*)d_out;

    // Workspace layout
    char* ws = (char*)d_ws;
    unsigned short* zbf = (unsigned short*)ws;                                   // 8 MB
    unsigned short* wbf = (unsigned short*)(ws + (size_t)BATCH * LATENT * 2);    // 2 MB
    float* wnorm   = (float*)(ws + (size_t)BATCH * LATENT * 2
                                 + (size_t)NUM_EMB * LATENT * 2);                // 32 KB
    float* partial = wnorm + NUM_EMB;                                            // 512 B

    float* out_zq   = out;                                  // [BATCH*LATENT]
    float* out_loss = out + (size_t)BATCH * LATENT;         // [2]
    float* out_idx  = out + (size_t)BATCH * LATENT + 2;     // [BATCH]

    vq_convert_z<<<(BATCH * LATENT) / 1024, 256, 0, stream>>>(z_e, zbf);
    vq_convert_w<<<NUM_EMB, 128, 0, stream>>>(W, wbf, wnorm);
    vq_main<<<BATCH / 256, 256, 0, stream>>>(zbf, wbf, wnorm, z_e, W,
                                             out_zq, out_idx, partial);
    vq_finalize<<<1, 32, 0, stream>>>(partial, out_loss);
}
